// MoELayer_21861383537416
// MI455X (gfx1250) — matrix operand order repaired and checked by reading, not compiled, not run on hardware
//
#include <hip/hip_runtime.h>
#include <hip/hip_bf16.h>
#include <math.h>

// ---------------- problem constants (match reference) ----------------
#define NTOK 4096   // B*S
#define DDIM 1024   // hidden
#define MDIM 2048   // mlp
#define NEXP 8      // experts
#define SIGMA_INV 8.0f   // 1/SIGMA, SIGMA = 1/E

// gfx1250 async global->LDS path (guarded: host pass and older toolchains
// fall back to synchronous vector copies)
#if defined(__has_builtin)
#  if __has_builtin(__builtin_amdgcn_global_load_async_to_lds_b128) && \
      __has_builtin(__builtin_amdgcn_s_wait_asynccnt)
#    define HAS_ASYNC 1
#  endif
#endif
#ifndef HAS_ASYNC
#  define HAS_ASYNC 0
#endif

typedef __attribute__((ext_vector_type(16))) __bf16 v16bf;
typedef __attribute__((ext_vector_type(8)))  __bf16 v8bf;
typedef __attribute__((ext_vector_type(8)))  float  v8f;

#if HAS_ASYNC
// b128 payload type expected by the async builtin (per compiler diagnostic:
// "int __vector(4) __device__ *")
typedef int v4i_b __attribute__((vector_size(16)));
typedef __attribute__((address_space(1))) v4i_b* gas_v4i;
typedef __attribute__((address_space(3))) v4i_b* las_v4i;
#endif

struct alignas(8) bf16x4 { __bf16 a, b, c, d; };

__device__ __forceinline__ v16bf pack16(v8bf lo, v8bf hi) {
    v16bf r;
#pragma unroll
    for (int i = 0; i < 8; ++i) { r[i] = lo[i]; r[i + 8] = hi[i]; }
    return r;
}

// A-fragment (16x32 bf16, M x K) from LDS tile stored [m][k], 32 k per row.
// lane<16: k in {0..7, 16..23}; lane>=16: k in {8..15, 24..31}
__device__ __forceinline__ v16bf load_afrag(const __bf16* As, int mt, int lane) {
    const __bf16* p = As + (mt * 16 + (lane & 15)) * 32 + ((lane >> 4) & 1) * 8;
    v8bf lo = *(const v8bf*)p;
    v8bf hi = *(const v8bf*)(p + 16);
    return pack16(lo, hi);
}

// B-fragment (32x16 bf16, K x N) from LDS tile stored transposed [n][k].
// lane<16: k in {0..7, 16..23}; lane>=16: k in {8..15, 24..31}
__device__ __forceinline__ v16bf load_bfrag(const __bf16* Bs, int nt, int lane) {
    const __bf16* p = Bs + (nt * 16 + (lane & 15)) * 32 + ((lane >> 4) & 1) * 8;
    v8bf lo = *(const v8bf*)p;
    v8bf hi = *(const v8bf*)(p + 16);
    return pack16(lo, hi);
}

__device__ __forceinline__ float gelu_exact(float v) {
    return 0.5f * v * (1.0f + erff(v * 0.70710678118654752f));
}

// Copy a 128-row x 32-col bf16 tile (8 KB) from global (row stride `stride`)
// into LDS laid out [row][32]. 16-byte chunks, 2 per thread, async if available.
__device__ __forceinline__ void stage_row_tile(const __bf16* __restrict__ g,
                                               size_t stride, __bf16* lds, int t) {
#pragma unroll
    for (int i = 0; i < 2; ++i) {
        int c = t + i * 256;
        int row = c >> 2, seg = (c & 3) * 8;
        const __bf16* gp = g + (size_t)row * stride + seg;
        __bf16* lp = lds + row * 32 + seg;
#if HAS_ASYNC
        __builtin_amdgcn_global_load_async_to_lds_b128(
            (gas_v4i)(__bf16*)gp, (las_v4i)lp, 0, 0);
#else
        *(v8bf*)lp = *(const v8bf*)gp;
#endif
    }
}

__device__ __forceinline__ void stage_wait() {
#if HAS_ASYNC
    __builtin_amdgcn_s_wait_asynccnt(0);
#endif
    __syncthreads();
}

// =====================================================================
// Pre-pass: elementwise f32 -> bf16 (for x)
// =====================================================================
__global__ __launch_bounds__(256) void convert_bf16_kernel(
    const float* __restrict__ in, __bf16* __restrict__ out, int n4)
{
    int i = blockIdx.x * 256 + threadIdx.x;
    if (i < n4) {
        float4 v = ((const float4*)in)[i];
        bf16x4 o;
        o.a = (__bf16)v.x; o.b = (__bf16)v.y; o.c = (__bf16)v.z; o.d = (__bf16)v.w;
        ((bf16x4*)out)[i] = o;
    }
}

// =====================================================================
// Pre-pass: f32 (K x N) -> bf16 transposed (N x K), 64x64 LDS tiles.
// blockIdx.z selects the matrix (expert); both src/dst are K*N apart.
// =====================================================================
__global__ __launch_bounds__(256) void transpose_convert_kernel(
    const float* __restrict__ in, __bf16* __restrict__ out, int K, int N)
{
    __shared__ __bf16 tile[64][65];
    const size_t mat = (size_t)K * N * blockIdx.z;
    const float* src = in + mat;
    __bf16* dst = out + mat;
    const int n0 = blockIdx.x * 64, k0 = blockIdx.y * 64;
    const int tx = threadIdx.x & 63, ty0 = threadIdx.x >> 6;
#pragma unroll
    for (int i = 0; i < 16; ++i) {
        int ty = ty0 + i * 4;
        tile[ty][tx] = (__bf16)src[(size_t)(k0 + ty) * N + n0 + tx];
    }
    __syncthreads();
#pragma unroll
    for (int i = 0; i < 16; ++i) {
        int ty = ty0 + i * 4;
        dst[(size_t)(n0 + ty) * K + k0 + tx] = tile[tx][ty];
    }
}

// =====================================================================
// Router: wave per token.
// =====================================================================
__global__ __launch_bounds__(256) void moe_router_kernel(
    const float* __restrict__ X, const float* __restrict__ GW,
    const float* __restrict__ noise,
    float* __restrict__ gating_out, float* __restrict__ logits_out,
    float* __restrict__ load_out, int* __restrict__ idx_out,
    float* __restrict__ combine_out)
{
    const int lane = threadIdx.x & 31;
    const int wave = threadIdx.x >> 5;
    const int token = blockIdx.x * 8 + wave;
    if (token >= NTOK) return;

    const float* xr = X + (size_t)token * DDIM;
    float s[NEXP];
#pragma unroll
    for (int e = 0; e < NEXP; ++e) s[e] = 0.0f;

    for (int j = lane; j < DDIM; j += 32) {
        float xv = xr[j];
#pragma unroll
        for (int e = 0; e < NEXP; ++e) s[e] += xv * GW[e * DDIM + j];
    }
#pragma unroll
    for (int off = 16; off > 0; off >>= 1) {
#pragma unroll
        for (int e = 0; e < NEXP; ++e) s[e] += __shfl_down(s[e], off, 32);
    }

    if (lane == 0) {
        float lg[NEXP], g[NEXP], nl[NEXP];
        float mx = s[0];
#pragma unroll
        for (int e = 0; e < NEXP; ++e) { lg[e] = s[e]; if (lg[e] > mx) mx = lg[e]; }
        float sum = 0.0f;
#pragma unroll
        for (int e = 0; e < NEXP; ++e) { g[e] = expf(lg[e] - mx); sum += g[e]; }
        float inv = 1.0f / sum;
#pragma unroll
        for (int e = 0; e < NEXP; ++e) {
            g[e] *= inv;
            logits_out[(size_t)token * NEXP + e] = lg[e];
            gating_out[(size_t)token * NEXP + e] = g[e];
        }
        int i1 = 0;
#pragma unroll
        for (int e = 1; e < NEXP; ++e) if (g[e] > g[i1]) i1 = e;
        int i2 = (i1 == 0) ? 1 : 0;
#pragma unroll
        for (int e = 0; e < NEXP; ++e) if (e != i1 && g[e] > g[i2]) i2 = e;
        float w1 = g[i1], w2 = g[i2];
        float den = 1.0f / (w1 + w2 + 1e-9f);
#pragma unroll
        for (int e = 0; e < NEXP; ++e) combine_out[(size_t)token * NEXP + e] = 0.0f;
        combine_out[(size_t)token * NEXP + i1] = w1 * den;
        combine_out[(size_t)token * NEXP + i2] = w2 * den;
        idx_out[(size_t)token * 2 + 0] = i1;
        idx_out[(size_t)token * 2 + 1] = i2;

#pragma unroll
        for (int e = 0; e < NEXP; ++e)
            nl[e] = lg[e] + noise[(size_t)token * NEXP + e] * (1.0f / SIGMA_INV);
        int j1 = 0;
#pragma unroll
        for (int e = 1; e < NEXP; ++e) if (nl[e] > nl[j1]) j1 = e;
        float tau = -3.4e38f;
#pragma unroll
        for (int e = 0; e < NEXP; ++e) if (e != j1 && nl[e] > tau) tau = nl[e];
#pragma unroll
        for (int e = 0; e < NEXP; ++e) {
            float z = (tau - lg[e]) * SIGMA_INV;
            load_out[(size_t)token * NEXP + e] = 0.5f * erfcf(z * 0.70710678118654752f);
        }
    }
}

// =====================================================================
// FC1: H = gelu(Xb @ W1t^T + b1), bf16 in/out, double-buffered async tiles
// block tile 128x128, K-step 32, 8 waves, wave tile 64x32 (4x2 WMMA tiles)
// =====================================================================
__global__ __launch_bounds__(256) void moe_fc1_kernel(
    const __bf16* __restrict__ Xb, const __bf16* __restrict__ W1t,
    const float* __restrict__ bias, __bf16* __restrict__ H)
{
    __shared__ __bf16 As[2][128 * 32];
    __shared__ __bf16 Bs[2][128 * 32];

    const int r0 = blockIdx.y * 128;      // token rows
    const int c0 = blockIdx.x * 128;      // mlp cols
    const int t = threadIdx.x;
    const int lane = t & 31, wave = t >> 5;
    const int mtb = (wave >> 2) * 4, ntb = (wave & 3) * 2;

    v8f acc[4][2];
#pragma unroll
    for (int mi = 0; mi < 4; ++mi)
#pragma unroll
        for (int ni = 0; ni < 2; ++ni)
#pragma unroll
            for (int r = 0; r < 8; ++r) acc[mi][ni][r] = 0.0f;

    const __bf16* Abase = Xb + (size_t)r0 * DDIM;
    const __bf16* Bbase = W1t + (size_t)c0 * DDIM;

    stage_row_tile(Abase, DDIM, As[0], t);
    stage_row_tile(Bbase, DDIM, Bs[0], t);
    stage_wait();

    const int NK = DDIM / 32;
    for (int ks = 0; ks < NK; ++ks) {
        const int cur = ks & 1;
        if (ks + 1 < NK) {
            stage_row_tile(Abase + (ks + 1) * 32, DDIM, As[cur ^ 1], t);
            stage_row_tile(Bbase + (ks + 1) * 32, DDIM, Bs[cur ^ 1], t);
        }
        v16bf af[4], bfr[2];
#pragma unroll
        for (int mi = 0; mi < 4; ++mi) af[mi] = load_afrag(As[cur], mtb + mi, lane);
#pragma unroll
        for (int ni = 0; ni < 2; ++ni) bfr[ni] = load_bfrag(Bs[cur], ntb + ni, lane);
#pragma unroll
        for (int mi = 0; mi < 4; ++mi)
#pragma unroll
            for (int ni = 0; ni < 2; ++ni)
                acc[mi][ni] = __builtin_amdgcn_wmma_f32_16x16x32_bf16(
                    false, af[mi], false, bfr[ni], (short)0, acc[mi][ni], false, false);
        stage_wait();
    }

    const int lrow = (lane & 16) ? 8 : 0;
#pragma unroll
    for (int mi = 0; mi < 4; ++mi)
#pragma unroll
        for (int ni = 0; ni < 2; ++ni) {
            int n = c0 + (ntb + ni) * 16 + (lane & 15);
            float bv = bias[n];
#pragma unroll
            for (int r = 0; r < 8; ++r) {
                int m = r0 + (mtb + mi) * 16 + r + lrow;
                float v = acc[mi][ni][r] + bv;
                H[(size_t)m * MDIM + n] = (__bf16)gelu_exact(v);
            }
        }
}

// =====================================================================
// FC2: out = (first ? 0 : out) + combine[:,e] * (H @ W2t^T + b2)
// =====================================================================
__global__ __launch_bounds__(256) void moe_fc2_kernel(
    const __bf16* __restrict__ H, const __bf16* __restrict__ W2t,
    const float* __restrict__ bias, const float* __restrict__ combine,
    int expert, int first, float* __restrict__ out)
{
    __shared__ __bf16 As[2][128 * 32];
    __shared__ __bf16 Bs[2][128 * 32];

    const int r0 = blockIdx.y * 128;      // token rows
    const int c0 = blockIdx.x * 128;      // hidden cols
    const int t = threadIdx.x;
    const int lane = t & 31, wave = t >> 5;
    const int mtb = (wave >> 2) * 4, ntb = (wave & 3) * 2;

    v8f acc[4][2];
#pragma unroll
    for (int mi = 0; mi < 4; ++mi)
#pragma unroll
        for (int ni = 0; ni < 2; ++ni)
#pragma unroll
            for (int r = 0; r < 8; ++r) acc[mi][ni][r] = 0.0f;

    const __bf16* Abase = H + (size_t)r0 * MDIM;
    const __bf16* Bbase = W2t + (size_t)c0 * MDIM;

    stage_row_tile(Abase, MDIM, As[0], t);
    stage_row_tile(Bbase, MDIM, Bs[0], t);
    stage_wait();

    const int NK = MDIM / 32;
    for (int ks = 0; ks < NK; ++ks) {
        const int cur = ks & 1;
        if (ks + 1 < NK) {
            stage_row_tile(Abase + (ks + 1) * 32, MDIM, As[cur ^ 1], t);
            stage_row_tile(Bbase + (ks + 1) * 32, MDIM, Bs[cur ^ 1], t);
        }
        v16bf af[4], bfr[2];
#pragma unroll
        for (int mi = 0; mi < 4; ++mi) af[mi] = load_afrag(As[cur], mtb + mi, lane);
#pragma unroll
        for (int ni = 0; ni < 2; ++ni) bfr[ni] = load_bfrag(Bs[cur], ntb + ni, lane);
#pragma unroll
        for (int mi = 0; mi < 4; ++mi)
#pragma unroll
            for (int ni = 0; ni < 2; ++ni)
                acc[mi][ni] = __builtin_amdgcn_wmma_f32_16x16x32_bf16(
                    false, af[mi], false, bfr[ni], (short)0, acc[mi][ni], false, false);
        stage_wait();
    }

    const int lrow = (lane & 16) ? 8 : 0;
#pragma unroll
    for (int mi = 0; mi < 4; ++mi)
#pragma unroll
        for (int ni = 0; ni < 2; ++ni) {
            int n = c0 + (ntb + ni) * 16 + (lane & 15);
            float bv = bias[n];
#pragma unroll
            for (int r = 0; r < 8; ++r) {
                int m = r0 + (mtb + mi) * 16 + r + lrow;
                float scale = combine[(size_t)m * NEXP + expert];
                float y = acc[mi][ni][r] + bv;
                size_t o = (size_t)m * DDIM + n;
                float prev = first ? 0.0f : out[o];
                out[o] = prev + scale * y;
            }
        }
}

// =====================================================================
extern "C" void kernel_launch(void* const* d_in, const int* in_sizes, int n_in,
                              void* d_out, int out_size, void* d_ws, size_t ws_size,
                              hipStream_t stream) {
    (void)in_sizes; (void)n_in; (void)out_size; (void)ws_size;
    const float* x      = (const float*)d_in[0];   // (B,S,D)
    const float* gate_w = (const float*)d_in[1];   // (E,D)
    const float* fc1_w  = (const float*)d_in[2];   // (E,D,M)
    const float* fc1_b  = (const float*)d_in[3];   // (E,M)
    const float* fc2_w  = (const float*)d_in[4];   // (E,M,D)
    const float* fc2_b  = (const float*)d_in[5];   // (E,D)
    const float* noise  = (const float*)d_in[6];   // (n,E)

    // outputs, concatenated flat in reference return order
    float* out_y      = (float*)d_out;                                  // n*D
    float* out_gating = out_y + (size_t)NTOK * DDIM;                    // n*E
    float* out_logits = out_gating + (size_t)NTOK * NEXP;               // n*E
    float* out_load   = out_logits + (size_t)NTOK * NEXP;               // n*E
    int*   out_idx    = (int*)(out_load + (size_t)NTOK * NEXP);         // n*K

    // workspace layout (bytes):
    //   combine : n*E f32                      (128 KB)
    //   Xb      : n*D bf16                     (8 MB)
    //   H       : n*M bf16                     (16 MB)
    //   W1t     : E*M*D bf16 (transposed)      (32 MB)
    //   W2t     : E*D*M bf16 (transposed)      (32 MB)
    char* ws = (char*)d_ws;
    float*  combine = (float*)ws;                         ws += (size_t)NTOK * NEXP * 4;
    __bf16* Xb      = (__bf16*)ws;                        ws += (size_t)NTOK * DDIM * 2;
    __bf16* Hbuf    = (__bf16*)ws;                        ws += (size_t)NTOK * MDIM * 2;
    __bf16* W1t     = (__bf16*)ws;                        ws += (size_t)NEXP * DDIM * MDIM * 2;
    __bf16* W2t     = (__bf16*)ws;

    dim3 blk(256);

    // pre-pass: convert x, transpose+convert weights to bf16 [n][k]
    const int xn4 = NTOK * DDIM / 4;
    convert_bf16_kernel<<<(xn4 + 255) / 256, blk, 0, stream>>>(x, Xb, xn4);
    transpose_convert_kernel<<<dim3(MDIM / 64, DDIM / 64, NEXP), blk, 0, stream>>>(
        fc1_w, W1t, DDIM, MDIM);
    transpose_convert_kernel<<<dim3(DDIM / 64, MDIM / 64, NEXP), blk, 0, stream>>>(
        fc2_w, W2t, MDIM, DDIM);

    moe_router_kernel<<<NTOK / 8, blk, 0, stream>>>(
        x, gate_w, noise, out_gating, out_logits, out_load, out_idx, combine);

    dim3 g1(MDIM / 128, NTOK / 128);   // (16, 32)
    dim3 g2(DDIM / 128, NTOK / 128);   // (8, 32)
    for (int e = 0; e < NEXP; ++e) {
        moe_fc1_kernel<<<g1, blk, 0, stream>>>(
            Xb, W1t + (size_t)e * DDIM * MDIM, fc1_b + (size_t)e * MDIM, Hbuf);
        moe_fc2_kernel<<<g2, blk, 0, stream>>>(
            Hbuf, W2t + (size_t)e * MDIM * DDIM, fc2_b + (size_t)e * DDIM,
            combine, e, (e == 0) ? 1 : 0, out_y);
    }
}
